// SelfAttention_59691455480591
// MI455X (gfx1250) — compile-verified, matrix operand order checked by reading, not run on hardware
//
#include <hip/hip_runtime.h>

// ---------------------------------------------------------------- constants
#define B_   4
#define S_   2048
#define D_   1024
#define H_   16
#define DH_  64
#define M_   (B_ * S_)          // 8192 rows of x

// ---------------------------------------------------------------- types
typedef __bf16 bf16_t;
typedef __attribute__((ext_vector_type(16))) __bf16          v16bf;
typedef __attribute__((ext_vector_type(8)))  float           v8f;
typedef __attribute__((ext_vector_type(8)))  unsigned short  u16x8;

union Frag16 {
    u16x8 h[2];
    v16bf bf;
};

// f32 -> bf16 round-to-nearest-even
__device__ __forceinline__ unsigned short f2bf(float f) {
    unsigned int u = __float_as_uint(f);
    u += 0x7FFFu + ((u >> 16) & 1u);
    return (unsigned short)(u >> 16);
}

// Load one 16x32 bf16 WMMA fragment from a row-major matrix (element (row,k)
// at base + row*ld + k).  Used for A-fragments (row = M) and B-fragments
// (row = N; all our GEMMs are X * W^T so K is contiguous for both operands).
// Layout per ISA 7.12.2: lane L<16 -> row L, K = {0..7, 16..23};
//                        lane L>=16 -> row L-16, K = {8..15, 24..31}.
__device__ __forceinline__ v16bf load_frag(const unsigned short* base, int ld,
                                           int row0, int k0) {
    const int lane = threadIdx.x & 31;
    const int r    = lane & 15;
    const int half = lane >> 4;
    const unsigned short* p = base + (size_t)(row0 + r) * (size_t)ld + k0 + half * 8;
    Frag16 f;
    f.h[0] = *(const u16x8*)(p);
    f.h[1] = *(const u16x8*)(p + 16);
    return f.bf;
}

#define WMMA_BF16(A, Bm, C) \
    __builtin_amdgcn_wmma_f32_16x16x32_bf16(false, (A), false, (Bm), (short)0, (C), false, false)

__device__ __forceinline__ v8f vzero8() {
    v8f z = {0.f, 0.f, 0.f, 0.f, 0.f, 0.f, 0.f, 0.f};
    return z;
}

// CDNA5 async copy: global -> LDS, 16 bytes per lane, tracked by ASYNCcnt.
// The LDS address operand is ptrtoint of the shared-memory destination: on
// amdgcn the low 32 bits of a generic LDS-aperture pointer ARE the LDS byte
// offset (ISA 10.2), and the ptrtoint escapes the shared object so the
// compiler must honor the asm's memory clobber (keeps the ds_load panel
// reads alive and ordered).
__device__ __forceinline__ void async_g2l_b128(unsigned short* lds_dst,
                                               const unsigned short* gaddr) {
    const unsigned lds_off = (unsigned)(size_t)(void*)lds_dst;
    asm volatile("global_load_async_to_lds_b128 %0, %1, off"
                 :: "v"(lds_off), "v"(gaddr)
                 : "memory");
}

__device__ __forceinline__ void wait_async0() {
    asm volatile("s_wait_asynccnt 0x0" ::: "memory");
}

// ---------------------------------------------------------------- kernel 1: casts
__global__ void cast_f32_to_bf16(const float* __restrict__ in,
                                 unsigned short* __restrict__ out, int n) {
    int i      = blockIdx.x * blockDim.x + threadIdx.x;
    int stride = gridDim.x * blockDim.x;
    for (; i < n; i += stride) out[i] = f2bf(in[i]);
}

// ---------------------------------------------------------------- GEMM core
// Block = 8 waves; covers 256 output rows x 64 output cols.
// Wave computes a 32x64 tile (2 m-tiles x 4 n-tiles, 8 WMMAs per K-step).
// B panel (64 rows of W x 32 K, 4KB bf16) staged in LDS, double buffered via
// async global->LDS DMA; A fragments double-buffered in registers.
__device__ __forceinline__ void gemm_core(const unsigned short* __restrict__ A,
                                          const unsigned short* __restrict__ W,
                                          int m0, int n0, int Kdim,
                                          unsigned short (*lds)[64 * 32],
                                          v8f acc[2][4]) {
    const int t     = threadIdx.x;
    const int row   = t >> 2;            // 0..63  (weight row within panel)
    const int chunk = t & 3;             // 4 x 16B chunks per 32-elem K row
    const unsigned short* g0 = W + (size_t)(n0 + row) * (size_t)Kdim + chunk * 8;
    const unsigned lds_elem  = (unsigned)(row * 32 + chunk * 8);

    // prologue: stage panel k0=0 into buffer 0, preload A fragments
    async_g2l_b128(&lds[0][lds_elem], g0);
    v16bf a0 = load_frag(A, Kdim, m0, 0);
    v16bf a1 = load_frag(A, Kdim, m0 + 16, 0);
    wait_async0();
    __syncthreads();

    int cur = 0;
    for (int k0 = 0; k0 < Kdim; k0 += 32) {
        // kick off DMA of next panel into the other buffer
        if (k0 + 32 < Kdim)
            async_g2l_b128(&lds[cur ^ 1][lds_elem], g0 + k0 + 32);

        // register-prefetch next A fragments (k clamped; extra load harmless)
        const int kn = (k0 + 32 < Kdim) ? (k0 + 32) : 0;
        v16bf a0n = load_frag(A, Kdim, m0, kn);
        v16bf a1n = load_frag(A, Kdim, m0 + 16, kn);

        // compute with current LDS panel (ds_load_b128 fragment reads)
        const unsigned short* bp = &lds[cur][0];
#pragma unroll
        for (int j = 0; j < 4; ++j) {
            v16bf b = load_frag(bp, 32, 16 * j, 0);
            acc[0][j] = WMMA_BF16(a0, b, acc[0][j]);
            acc[1][j] = WMMA_BF16(a1, b, acc[1][j]);
        }

        wait_async0();       // next panel resident (this wave's slice)
        __syncthreads();     // everyone staged & done reading current
        cur ^= 1;
        a0 = a0n;
        a1 = a1n;
    }
}

// ---------------------------------------------------------------- kernel 2: QKV GEMM
// qkv = x @ w_in^T + b_in ; scatter into Q [B,H,S,Dh] (scaled by 1/8),
// K [B,H,S,Dh], V transposed [B,H,Dh,S].
__global__ void __launch_bounds__(256)
qkv_gemm_wmma(const unsigned short* __restrict__ xb,   // [M_, D_] bf16
              const unsigned short* __restrict__ wb,   // [3D_, D_] bf16
              const float*          __restrict__ b_in, // [3D_]
              unsigned short* __restrict__ Q,
              unsigned short* __restrict__ Kk,
              unsigned short* __restrict__ Vt) {
    __shared__ __align__(16) unsigned short lds[2][64 * 32];   // 8KB

    const int NG   = (3 * D_) / 64;            // 48 col groups
    const int mblk = blockIdx.x / NG;
    const int nblk = blockIdx.x % NG;
    const int wave = threadIdx.x >> 5;
    const int m0   = mblk * 256 + wave * 32;
    const int n0   = nblk * 64;
    const int lane = threadIdx.x & 31;
    const int nl   = lane & 15;
    const int half = lane >> 4;

    v8f acc[2][4];
#pragma unroll
    for (int ti = 0; ti < 2; ++ti)
#pragma unroll
        for (int j = 0; j < 4; ++j) acc[ti][j] = vzero8();

    gemm_core(xb, wb, m0, n0, D_, lds, acc);

#pragma unroll
    for (int ti = 0; ti < 2; ++ti) {
#pragma unroll
        for (int j = 0; j < 4; ++j) {
            const int   col  = n0 + 16 * j + nl;
            const int   part = col >> 10;          // 0=q 1=k 2=v
            const int   d    = col & (D_ - 1);
            const int   h    = d >> 6;
            const int   dh   = d & (DH_ - 1);
            const float bias = b_in[col];
#pragma unroll
            for (int r = 0; r < 8; ++r) {
                const int   m  = m0 + ti * 16 + r + 8 * half;
                const int   b  = m >> 11;          // / S_
                const int   s  = m & (S_ - 1);
                const float v  = acc[ti][j][r] + bias;
                const size_t hs = ((size_t)(b * H_ + h) * S_ + s) * DH_ + dh;
                if (part == 0) {
                    Q[hs] = f2bf(v * 0.125f);       // fold 1/sqrt(Dh)
                } else if (part == 1) {
                    Kk[hs] = f2bf(v);
                } else {
                    Vt[((size_t)(b * H_ + h) * DH_ + dh) * S_ + s] = f2bf(v);
                }
            }
        }
    }
}

// ---------------------------------------------------------------- kernel 3: flash attention
// One wave owns a 16-row q strip; block of 8 waves covers 128 q rows of one
// (b,h).  Per-wave LDS tile transposes P from C-layout to A-layout
// (intra-wave only -> no barriers; waves have different causal trip counts).
__global__ void __launch_bounds__(256)
attn_wmma(const unsigned short* __restrict__ Q,
          const unsigned short* __restrict__ Kk,
          const unsigned short* __restrict__ Vt,
          unsigned short* __restrict__ ctx) {            // [M_, D_] bf16
    __shared__ __align__(16) unsigned short pl[8][16 * 32];

    const int wave = threadIdx.x >> 5;
    const int lane = threadIdx.x & 31;
    const int nl   = lane & 15;
    const int half = lane >> 4;
    const int bh   = blockIdx.x >> 4;       // 0..63
    const int qt   = blockIdx.x & 15;
    const int q0   = qt * 128 + wave * 16;
    const int b    = bh >> 4;
    const int h    = bh & 15;

    const unsigned short* Qh = Q  + (size_t)bh * S_ * DH_;
    const unsigned short* Kh = Kk + (size_t)bh * S_ * DH_;
    const unsigned short* Vh = Vt + (size_t)bh * DH_ * S_;

    const v16bf aq0 = load_frag(Qh, DH_, q0, 0);
    const v16bf aq1 = load_frag(Qh, DH_, q0, 32);

    v8f mrow, lrow, o[4];
#pragma unroll
    for (int r = 0; r < 8; ++r) { mrow[r] = -1e30f; lrow[r] = 0.f; }
#pragma unroll
    for (int j = 0; j < 4; ++j) o[j] = vzero8();

    unsigned short* pw = &pl[wave][0];

    for (int k0 = 0; k0 < q0 + 16; k0 += 32) {
        // ---- scores: 16 q rows x 32 keys = two 16x16 C-fragments
        v8f s0 = vzero8(), s1 = vzero8();
        {
            v16bf bk;
            bk = load_frag(Kh, DH_, k0,       0);  s0 = WMMA_BF16(aq0, bk, s0);
            bk = load_frag(Kh, DH_, k0,      32);  s0 = WMMA_BF16(aq1, bk, s0);
            bk = load_frag(Kh, DH_, k0 + 16,  0);  s1 = WMMA_BF16(aq0, bk, s1);
            bk = load_frag(Kh, DH_, k0 + 16, 32);  s1 = WMMA_BF16(aq1, bk, s1);
        }

        // ---- hoist V B-fragments: their global loads overlap softmax VALU
        v16bf bv[4];
#pragma unroll
        for (int j = 0; j < 4; ++j) bv[j] = load_frag(Vh, S_, j * 16, k0);

        // ---- causal mask + online softmax in C-layout
        const int kg0 = k0 + nl;
        const int kg1 = k0 + 16 + nl;
        v8f p0, p1, scl;
#pragma unroll
        for (int r = 0; r < 8; ++r) {
            const int qg = q0 + r + 8 * half;
            float v0 = (kg0 <= qg) ? s0[r] : -1e30f;
            float v1 = (kg1 <= qg) ? s1[r] : -1e30f;
            float mx = fmaxf(v0, v1);
            mx = fmaxf(mx, __shfl_xor(mx, 1, 32));
            mx = fmaxf(mx, __shfl_xor(mx, 2, 32));
            mx = fmaxf(mx, __shfl_xor(mx, 4, 32));
            mx = fmaxf(mx, __shfl_xor(mx, 8, 32));
            const float mn = fmaxf(mrow[r], mx);
            const float e0 = __expf(v0 - mn);
            const float e1 = __expf(v1 - mn);
            const float sc = __expf(mrow[r] - mn);
            float sum = e0 + e1;
            sum += __shfl_xor(sum, 1, 32);
            sum += __shfl_xor(sum, 2, 32);
            sum += __shfl_xor(sum, 4, 32);
            sum += __shfl_xor(sum, 8, 32);
            lrow[r] = lrow[r] * sc + sum;
            mrow[r] = mn;
            p0[r] = e0; p1[r] = e1; scl[r] = sc;
        }
#pragma unroll
        for (int j = 0; j < 4; ++j)
#pragma unroll
            for (int r = 0; r < 8; ++r) o[j][r] *= scl[r];

        // ---- C-layout P -> LDS -> A-layout P (intra-wave transpose)
#pragma unroll
        for (int r = 0; r < 8; ++r) {
            const int row = r + 8 * half;
            pw[row * 32 + nl]      = f2bf(p0[r]);
            pw[row * 32 + 16 + nl] = f2bf(p1[r]);
        }
        const v16bf pa = load_frag(pw, 32, 0, 0);

        // ---- O += P @ V  (V pre-transposed: row = dv, contiguous in key idx)
#pragma unroll
        for (int j = 0; j < 4; ++j) o[j] = WMMA_BF16(pa, bv[j], o[j]);
    }

    // ---- epilogue: O / l, scatter into [b, s, h, dv] = [M_, D_]
    v8f inv;
#pragma unroll
    for (int r = 0; r < 8; ++r) inv[r] = 1.0f / lrow[r];
#pragma unroll
    for (int j = 0; j < 4; ++j) {
        const int dv = j * 16 + nl;
#pragma unroll
        for (int r = 0; r < 8; ++r) {
            const int qg = q0 + r + 8 * half;
            ctx[(size_t)(b * S_ + qg) * D_ + h * DH_ + dv] = f2bf(o[j][r] * inv[r]);
        }
    }
}

// ---------------------------------------------------------------- kernel 4: output GEMM
// out = ctx @ w_out^T + b_out  (f32 result)
__global__ void __launch_bounds__(256)
oproj_gemm_wmma(const unsigned short* __restrict__ ctx,  // [M_, D_] bf16
                const unsigned short* __restrict__ wob,  // [D_, D_] bf16
                const float*          __restrict__ b_out,
                float* __restrict__ out) {               // [M_, D_] f32
    __shared__ __align__(16) unsigned short lds[2][64 * 32];   // 8KB

    const int NG   = D_ / 64;                 // 16 col groups
    const int mblk = blockIdx.x / NG;
    const int nblk = blockIdx.x % NG;
    const int wave = threadIdx.x >> 5;
    const int m0   = mblk * 256 + wave * 32;
    const int n0   = nblk * 64;
    const int lane = threadIdx.x & 31;
    const int nl   = lane & 15;
    const int half = lane >> 4;

    v8f acc[2][4];
#pragma unroll
    for (int ti = 0; ti < 2; ++ti)
#pragma unroll
        for (int j = 0; j < 4; ++j) acc[ti][j] = vzero8();

    gemm_core(ctx, wob, m0, n0, D_, lds, acc);

#pragma unroll
    for (int ti = 0; ti < 2; ++ti) {
#pragma unroll
        for (int j = 0; j < 4; ++j) {
            const int   col  = n0 + 16 * j + nl;
            const float bias = b_out[col];
#pragma unroll
            for (int r = 0; r < 8; ++r) {
                const int m = m0 + ti * 16 + r + 8 * half;
                out[(size_t)m * D_ + col] = acc[ti][j][r] + bias;
            }
        }
    }
}

// ---------------------------------------------------------------- launcher
extern "C" void kernel_launch(void* const* d_in, const int* in_sizes, int n_in,
                              void* d_out, int out_size, void* d_ws, size_t ws_size,
                              hipStream_t stream) {
    const float* x     = (const float*)d_in[0];   // [4,2048,1024]
    const float* w_in  = (const float*)d_in[1];   // [3072,1024]
    const float* b_in  = (const float*)d_in[2];   // [3072]
    const float* w_out = (const float*)d_in[3];   // [1024,1024]
    const float* b_out = (const float*)d_in[4];   // [1024]
    float*       out   = (float*)d_out;           // [4,2048,1024]

    // workspace layout (bf16 elements)
    unsigned short* xb  = (unsigned short*)d_ws;                 // M_*D_
    unsigned short* wib = xb  + (size_t)M_ * D_;                 // 3*D_*D_
    unsigned short* wob = wib + (size_t)3 * D_ * D_;             // D_*D_
    unsigned short* Q   = wob + (size_t)D_ * D_;                 // M_*D_
    unsigned short* Kk  = Q   + (size_t)M_ * D_;                 // M_*D_
    unsigned short* Vt  = Kk  + (size_t)M_ * D_;                 // M_*D_
    unsigned short* ctx = Vt  + (size_t)M_ * D_;                 // M_*D_

    cast_f32_to_bf16<<<2048, 256, 0, stream>>>(x,     xb,  M_ * D_);
    cast_f32_to_bf16<<<1024, 256, 0, stream>>>(w_in,  wib, 3 * D_ * D_);
    cast_f32_to_bf16<<<512,  256, 0, stream>>>(w_out, wob, D_ * D_);

    // QKV: 32 m-blocks (256 rows) x 48 n-groups (64 cols) = 1536 blocks
    qkv_gemm_wmma<<<1536, 256, 0, stream>>>(xb, wib, b_in, Q, Kk, Vt);

    // Attention: 64 (b,h) x 16 q-tiles of 128 rows = 1024 blocks
    attn_wmma<<<1024, 256, 0, stream>>>(Q, Kk, Vt, ctx);

    // Output projection: 32 m-blocks x 16 n-groups = 512 blocks
    oproj_gemm_wmma<<<512, 256, 0, stream>>>(ctx, wob, b_out, out);
}
